// SwitchGNN_54631984005526
// MI455X (gfx1250) — matrix operand order repaired and checked by reading, not compiled, not run on hardware
//
#include <hip/hip_runtime.h>

// SwitchGNN for MI455X (gfx1250, wave32).
// Structure: for each of 8 passes, (1) WMMA bf16x3 GEMM H = x @ W[t],
// (2) warp-per-edge atomic scatter-mean, (3) accumulate mean/8 into out.
// H (51MB) + acc (51MB) stay L2-resident (192MB L2) -> scatter phase runs
// at L2 bandwidth. GEMM uses v_wmma_f32_16x16x32_bf16 with a 3-term
// hi/lo split for ~fp32 accuracy.

typedef __attribute__((ext_vector_type(16))) __bf16 v16bf;
typedef __attribute__((ext_vector_type(8)))  float  v8f;

#define N_NODES 100000
#define DIM 128
#define E_PER_TYPE 250000

// ---------------------------------------------------------------------------
// GEMM: H[nrows, 128] = X[nrows, 128] @ W[128, 128], bf16x3 split via WMMA.
// Block = 256 threads = 8 waves; each wave owns a 16-row strip, block covers
// 128 rows x 128 cols. W is converted to bf16 hi/lo once per block and staged
// in LDS already swizzled into per-lane WMMA B-fragment order:
//   frag(kc, ntile)[lane][j]  at  lds[(((kc*8+ntile)*32)+lane)*16 + j]
// B layout (16x16x32 bf16): lanes 0-15 = N 0-15 holding K {0-7,16-23},
// lanes 16-31 = same N holding K {8-15,24-31}.
// A layout: lanes 0-15 = rows M 0-15 holding K {0-7,16-23},
// lanes 16-31 = same rows holding K {8-15,24-31}.
// ---------------------------------------------------------------------------
__global__ __launch_bounds__(256) void gemm_bf16x3_kernel(
    const float* __restrict__ X, const float* __restrict__ W,
    float* __restrict__ H, int nrows) {
  __shared__ __bf16 lds_hi[4 * 8 * 32 * 16];  // 16384 elems = 32 KB
  __shared__ __bf16 lds_lo[4 * 8 * 32 * 16];  // 32 KB

  const int tid = threadIdx.x;

  // Cooperative W -> bf16 hi/lo split, stored in WMMA B-fragment order.
  for (int idx = tid; idx < 16384; idx += 256) {
    const int j    = idx & 15;
    const int L    = (idx >> 4) & 31;
    const int nt   = (idx >> 9) & 7;
    const int kc   = idx >> 12;
    const int half = L >> 4;
    const int lin  = L & 15;
    const int K    = kc * 32 + (j < 8 ? half * 8 + j : 16 + half * 8 + (j - 8));  // B: elem j <-> K = half*8 + j (j<8), 16 + half*8 + j-8 (j>=8)
    const int Ncol = nt * 16 + lin;
    const float w  = W[K * DIM + Ncol];
    const __bf16 h = (__bf16)w;
    lds_hi[idx] = h;
    lds_lo[idx] = (__bf16)(w - (float)h);
  }
  __syncthreads();

  const int wave = tid >> 5;
  const int lane = tid & 31;
  const int half = lane >> 4;
  const int lin  = lane & 15;

  const int rbase = blockIdx.x * 128 + wave * 16;
  int arow = rbase + lin;                       // A-load row (both halves)
  if (arow >= nrows) arow = nrows - 1;          // clamp (stores are masked)
  const float* __restrict__ xrow = X + (size_t)arow * DIM;

  v8f acc[8];
#pragma unroll
  for (int n = 0; n < 8; ++n) acc[n] = v8f{0.f, 0.f, 0.f, 0.f, 0.f, 0.f, 0.f, 0.f};

#pragma unroll
  for (int kc = 0; kc < 4; ++kc) {
    // A fragment: two contiguous 8-float groups per lane, split hi/lo.
    v16bf a_hi, a_lo;
    const float* g0 = xrow + kc * 32 + half * 8;       // elems 0..7
    const float* g1 = xrow + kc * 32 + 16 + half * 8;  // elems 8..15
#pragma unroll
    for (int j = 0; j < 8; ++j) {
      const float v0 = g0[j];
      const float v1 = g1[j];
      const __bf16 h0 = (__bf16)v0;
      const __bf16 h1 = (__bf16)v1;
      a_hi[j]     = h0;  a_lo[j]     = (__bf16)(v0 - (float)h0);
      a_hi[j + 8] = h1;  a_lo[j + 8] = (__bf16)(v1 - (float)h1);
    }

#pragma unroll
    for (int nt = 0; nt < 8; ++nt) {
      const int fbase = (((kc * 8 + nt) * 32) + lane) * 16;
      const v16bf b_hi = *reinterpret_cast<const v16bf*>(&lds_hi[fbase]);
      const v16bf b_lo = *reinterpret_cast<const v16bf*>(&lds_lo[fbase]);
      // C += Ahi*Bhi + Ahi*Blo + Alo*Bhi  (~fp32-class accuracy)
      acc[nt] = __builtin_amdgcn_wmma_f32_16x16x32_bf16(
          false, a_hi, false, b_hi, (short)0, acc[nt], false, false);
      acc[nt] = __builtin_amdgcn_wmma_f32_16x16x32_bf16(
          false, a_hi, false, b_lo, (short)0, acc[nt], false, false);
      acc[nt] = __builtin_amdgcn_wmma_f32_16x16x32_bf16(
          false, a_lo, false, b_hi, (short)0, acc[nt], false, false);
    }
  }

  // C/D layout: lane half h, VGPR r -> row rbase + h*8 + r, col nt*16 + lin.
#pragma unroll
  for (int nt = 0; nt < 8; ++nt) {
#pragma unroll
    for (int r = 0; r < 8; ++r) {
      const int m = rbase + half * 8 + r;
      if (m < nrows) H[(size_t)m * DIM + nt * 16 + lin] = acc[nt][r];
    }
  }
}

// ---------------------------------------------------------------------------
// Warp-per-edge scatter: 32 lanes x float4 = 128 floats per message.
// acc[d] += H[s]; cnt[d] += 1. H/acc are L2-resident.
// ---------------------------------------------------------------------------
__global__ __launch_bounds__(256) void scatter_kernel(
    const float* __restrict__ H, const int* __restrict__ edges, int nedges,
    float* __restrict__ acc, int* __restrict__ cnt) {
  const int warp = (blockIdx.x * blockDim.x + threadIdx.x) >> 5;
  const int lane = threadIdx.x & 31;
  if (warp >= nedges) return;
  const int s = edges[warp];           // e[0][i] : gather source
  const int d = edges[nedges + warp];  // e[1][i] : scatter destination
  const float4 v =
      *(reinterpret_cast<const float4*>(H + (size_t)s * DIM) + lane);
  float* dst = acc + (size_t)d * DIM + lane * 4;
  atomicAdd(dst + 0, v.x);
  atomicAdd(dst + 1, v.y);
  atomicAdd(dst + 2, v.z);
  atomicAdd(dst + 3, v.w);
  if (lane == 0) atomicAdd(cnt + d, 1);
}

// out += (acc / max(cnt,1)) * (1/8)
__global__ __launch_bounds__(256) void finalize_kernel(
    const float* __restrict__ acc, const int* __restrict__ cnt,
    float* __restrict__ out, int total) {
  const int i = blockIdx.x * blockDim.x + threadIdx.x;
  if (i >= total) return;
  float c = (float)cnt[i >> 7];  // i / DIM
  c = c < 1.f ? 1.f : c;
  out[i] += acc[i] * (0.125f / c);
}

extern "C" void kernel_launch(void* const* d_in, const int* in_sizes, int n_in,
                              void* d_out, int out_size, void* d_ws,
                              size_t ws_size, hipStream_t stream) {
  const float* x = (const float*)d_in[0];             // [100000,128] f32
  const float* W = (const float*)d_in[1];             // [8,128,128]  f32
  const int* e[7];
  for (int t = 0; t < 7; ++t) e[t] = (const int*)d_in[2 + t];  // [2,E] i32
  const int nedges = in_sizes[2] / 2;                 // 250000

  float* out = (float*)d_out;

  // Workspace: H (51.2MB) | acc (51.2MB) | cnt (0.4MB)  -> ~103MB total.
  float* H   = (float*)d_ws;
  float* acc = H + (size_t)N_NODES * DIM;
  int*   cnt = (int*)(acc + (size_t)N_NODES * DIM);

  hipMemsetAsync(out, 0, sizeof(float) * (size_t)N_NODES * DIM, stream);

  const dim3 gemmGrid((N_NODES + 127) / 128);
  const dim3 blk(256);
  const dim3 scatGrid((nedges * 32 + 255) / 256);
  const dim3 finGrid((N_NODES * DIM + 255) / 256);

  for (int t = 0; t < 8; ++t) {
    gemm_bf16x3_kernel<<<gemmGrid, blk, 0, stream>>>(
        x, W + (size_t)t * DIM * DIM, H, N_NODES);
    hipMemsetAsync(acc, 0, sizeof(float) * (size_t)N_NODES * DIM, stream);
    hipMemsetAsync(cnt, 0, sizeof(int) * N_NODES, stream);
    if (t < 7) {
      scatter_kernel<<<scatGrid, blk, 0, stream>>>(H, e[t], nedges, acc, cnt);
    } else {
      // Global pass: all edges, messages from H = x @ W[7].
      for (int u = 0; u < 7; ++u)
        scatter_kernel<<<scatGrid, blk, 0, stream>>>(H, e[u], nedges, acc, cnt);
    }
    finalize_kernel<<<finGrid, blk, 0, stream>>>(acc, cnt, out, N_NODES * DIM);
  }
}
